// VQ_24781961298653
// MI455X (gfx1250) — matrix operand order repaired and checked by reading, not compiled, not run on hardware
//
#include <hip/hip_runtime.h>
#include <math.h>

typedef __bf16 bf16_t;
typedef bf16_t v16bf __attribute__((ext_vector_type(16)));
typedef bf16_t v8bf  __attribute__((ext_vector_type(8)));
typedef bf16_t v4bf  __attribute__((ext_vector_type(4)));
typedef float  v8f   __attribute__((ext_vector_type(8)));

#define VQ_D     512
#define VQ_K     2048
#define VQ_N     16384
#define VQ_ND    (VQ_N * VQ_D)
#define ROWS_PB  64        // rows per block (A tile in LDS)
#define COLS_PI  32        // codes per c0 group (2 col-tiles x 16)
#define KCHUNK   64        // staged K-chunk of the codebook
#define NCHUNK   ((VQ_K / COLS_PI) * (VQ_D / KCHUNK))   // 512 chunks total

struct bfpair { bf16_t h; bf16_t l; };

// Split an fp32 into bf16 hi (exact truncation) + bf16 lo (residual, RN).
__device__ __forceinline__ bfpair splitbf(float x) {
    unsigned ux = __float_as_uint(x);
    float fh = __uint_as_float(ux & 0xffff0000u);
    bfpair r;
    r.h = (bf16_t)fh;            // exact: fh has <=8 mantissa bits
    r.l = (bf16_t)(x - fh);      // next ~8 bits of mantissa
    return r;
}

// Workgroup split-barrier that drains ONLY the DS counter: LDS stores become
// visible, but prefetched global loads may stay in flight across the barrier
// (they complete under the WMMA stream; __syncthreads would drain LOADcnt too).
__device__ __forceinline__ void lds_barrier() {
    asm volatile(
        "s_wait_dscnt 0x0\n\t"
        "s_barrier_signal -1\n\t"
        "s_barrier_wait -1"
        ::: "memory");
}

// ---------------------------------------------------------------------------
// Kernel 1: row norms (fp32, exact) + zero counts/sse accumulators.
// ---------------------------------------------------------------------------
__global__ void vq_norms_init_kernel(const float* __restrict__ X,
                                     const float* __restrict__ C,
                                     float* __restrict__ x2,
                                     float* __restrict__ c2,
                                     float* __restrict__ counts_sse) {
    if (blockIdx.x == 0) {
        for (int i = threadIdx.x; i < VQ_K + 1; i += blockDim.x)
            counts_sse[i] = 0.0f;
    }
    const int gid = blockIdx.x * 16 + (threadIdx.x >> 4);
    const int t   = threadIdx.x & 15;
    const float* row; float* dst;
    if (gid < VQ_N) { row = X + (size_t)gid * VQ_D;          dst = x2 + gid; }
    else            { row = C + (size_t)(gid - VQ_N) * VQ_D; dst = c2 + (gid - VQ_N); }
    float s = 0.0f;
    for (int k = t; k < VQ_D; k += 16) { float v = row[k]; s += v * v; }
    #pragma unroll
    for (int off = 8; off >= 1; off >>= 1) s += __shfl_xor(s, off, 16);
    if (t == 0) *dst = s;
}

// ---------------------------------------------------------------------------
// Kernel 2: split-bf16 WMMA distance GEMM + fused argmin.
// Block: 256 threads = 8 waves = 4 row-tiles x 2 col-tiles; 64 rows/block.
// A (64x512) staged once in LDS as bf16 hi/lo planes. Codebook staged in
// 32x64 chunks with double buffering + register prefetch; DS-only split
// barrier per chunk; 6x v_wmma_f32_16x16x32_bf16 per chunk.
// ---------------------------------------------------------------------------
__global__ void __launch_bounds__(256)
vq_argmin_kernel(const float* __restrict__ X,
                 const float* __restrict__ C,
                 const float* __restrict__ x2,
                 const float* __restrict__ c2,
                 int* __restrict__ out_idx) {
    __shared__ bf16_t Ahi[ROWS_PB * VQ_D];        // 64 KB
    __shared__ bf16_t Alo[ROWS_PB * VQ_D];        // 64 KB
    __shared__ bf16_t Bhi[2][COLS_PI * KCHUNK];   // 2 x 4 KB
    __shared__ bf16_t Blo[2][COLS_PI * KCHUNK];   // 2 x 4 KB
    __shared__ float  sBD[8][16];
    __shared__ int    sBI[8][16];

    const int tid  = threadIdx.x;
    const int wave = tid >> 5;
    const int lane = tid & 31;
    const int h    = lane >> 4;     // K-half select per ISA 16-bit layouts
    const int l    = lane & 15;     // M (for A) / N (for B) index
    const int rt   = wave >> 1;     // row subtile 0..3
    const int ct   = wave & 1;      // col subtile 0..1
    const int rowBase = blockIdx.x * ROWS_PB;

    // ---- stage A tile: fp32 -> split bf16, row-major planes ----
    for (int e = tid * 4; e < ROWS_PB * VQ_D; e += 256 * 4) {
        float4 xv = *(const float4*)(X + (size_t)rowBase * VQ_D + e);
        v4bf hv, lv;
        bfpair p0 = splitbf(xv.x); hv[0] = p0.h; lv[0] = p0.l;
        bfpair p1 = splitbf(xv.y); hv[1] = p1.h; lv[1] = p1.l;
        bfpair p2 = splitbf(xv.z); hv[2] = p2.h; lv[2] = p2.l;
        bfpair p3 = splitbf(xv.w); hv[3] = p3.h; lv[3] = p3.l;
        *(v4bf*)(Ahi + e) = hv;
        *(v4bf*)(Alo + e) = lv;
    }

    // Per-lane row norms: acc element v maps to row rt*16 + v + 8*h.
    float x2r[8];
    #pragma unroll
    for (int v = 0; v < 8; ++v) x2r[v] = x2[rowBase + rt * 16 + 8 * h + v];

    float bestD[8]; int bestI[8];
    #pragma unroll
    for (int v = 0; v < 8; ++v) { bestD[v] = 3.402823466e+38f; bestI[v] = 0; }

    const bf16_t* aRowHi = Ahi + (rt * 16 + l) * VQ_D;
    const bf16_t* aRowLo = Alo + (rt * 16 + l) * VQ_D;

    // Staging geometry: 32 codes x 64 ks = 2048 elems, 8 per thread.
    const int sCol = tid >> 3;            // code within group
    const int sK   = (tid & 7) * 8;       // k offset within chunk
    const int eSt  = tid * 8;             // LDS element offset

    // Prefetch chunk 0 into registers.
    float4 g0, g1;
    {
        const float* src = C + (size_t)sCol * VQ_D + sK;
        g0 = *(const float4*)(src);
        g1 = *(const float4*)(src + 4);
    }

    v8f acc = {0.f, 0.f, 0.f, 0.f, 0.f, 0.f, 0.f, 0.f};

    for (int t = 0; t < NCHUNK; ++t) {
        const int c0  = (t >> 3) * COLS_PI;
        const int kk  = (t & 7) * KCHUNK;
        const int buf = t & 1;

        // ---- convert + store the prefetched chunk into buffer `buf` ----
        {
            v8bf hv, lv;
            bfpair q0 = splitbf(g0.x); hv[0] = q0.h; lv[0] = q0.l;
            bfpair q1 = splitbf(g0.y); hv[1] = q1.h; lv[1] = q1.l;
            bfpair q2 = splitbf(g0.z); hv[2] = q2.h; lv[2] = q2.l;
            bfpair q3 = splitbf(g0.w); hv[3] = q3.h; lv[3] = q3.l;
            bfpair q4 = splitbf(g1.x); hv[4] = q4.h; lv[4] = q4.l;
            bfpair q5 = splitbf(g1.y); hv[5] = q5.h; lv[5] = q5.l;
            bfpair q6 = splitbf(g1.z); hv[6] = q6.h; lv[6] = q6.l;
            bfpair q7 = splitbf(g1.w); hv[7] = q7.h; lv[7] = q7.l;
            *(v8bf*)(&Bhi[buf][0] + eSt) = hv;
            *(v8bf*)(&Blo[buf][0] + eSt) = lv;
        }

        // ---- prefetch next chunk; stays in flight across the barrier ----
        if (t + 1 < NCHUNK) {
            const int tn  = t + 1;
            const int nc0 = (tn >> 3) * COLS_PI;
            const int nkk = (tn & 7) * KCHUNK;
            const float* src = C + (size_t)(nc0 + sCol) * VQ_D + nkk + sK;
            g0 = *(const float4*)(src);
            g1 = *(const float4*)(src + 4);
        }

        lds_barrier();   // drains DScnt only; prefetch loads keep flying

        // ---- 2 sub-blocks of K=32: 3 bf16 WMMAs each ----
        #pragma unroll
        for (int s = 0; s < 2; ++s) {
            const int kb = kk + s * 32;
            v8bf a0 = *(const v8bf*)(aRowHi + kb + h * 8);
            v8bf a1 = *(const v8bf*)(aRowHi + kb + 16 + h * 8);
            v16bf Ah = __builtin_shufflevector(a0, a1,
                0,1,2,3,4,5,6,7,8,9,10,11,12,13,14,15);
            v8bf a2 = *(const v8bf*)(aRowLo + kb + h * 8);
            v8bf a3 = *(const v8bf*)(aRowLo + kb + 16 + h * 8);
            v16bf Al = __builtin_shufflevector(a2, a3,
                0,1,2,3,4,5,6,7,8,9,10,11,12,13,14,15);

            const bf16_t* bh = &Bhi[buf][0] + (ct * 16 + l) * KCHUNK + s * 32 + h * 8;
            const bf16_t* bl = &Blo[buf][0] + (ct * 16 + l) * KCHUNK + s * 32 + h * 8;
            v8bf b0 = *(const v8bf*)(bh);
            v8bf b1 = *(const v8bf*)(bh + 16);
            v16bf Bh = __builtin_shufflevector(b0, b1,
                0,1,2,3,4,5,6,7,8,9,10,11,12,13,14,15);
            v8bf b2 = *(const v8bf*)(bl);
            v8bf b3 = *(const v8bf*)(bl + 16);
            v16bf Bl = __builtin_shufflevector(b2, b3,
                0,1,2,3,4,5,6,7,8,9,10,11,12,13,14,15);

            acc = __builtin_amdgcn_wmma_f32_16x16x32_bf16(
                false, Ah, false, Bh, (short)0, acc, false, false);
            acc = __builtin_amdgcn_wmma_f32_16x16x32_bf16(
                false, Ah, false, Bl, (short)0, acc, false, false);
            acc = __builtin_amdgcn_wmma_f32_16x16x32_bf16(
                false, Al, false, Bh, (short)0, acc, false, false);
        }

        // ---- end of a c0 group: distances + branchless argmin ----
        if ((t & 7) == 7) {
            const float c2n = c2[c0 + ct * 16 + l];
            #pragma unroll
            for (int v = 0; v < 8; ++v) {
                float d = x2r[v] + c2n - 2.0f * acc[v];
                int   n = c0 + ct * 16 + l;
                #pragma unroll
                for (int off = 8; off >= 1; off >>= 1) {
                    float od = __shfl_xor(d, off, 16);
                    int   oi = __shfl_xor(n, off, 16);
                    int take = (int)(od < d) | ((int)(od == d) & (int)(oi < n));
                    d = take ? od : d;
                    n = take ? oi : n;
                }
                int upd = d < bestD[v];
                bestD[v] = upd ? d : bestD[v];
                bestI[v] = upd ? n : bestI[v];
            }
            acc = (v8f){0.f, 0.f, 0.f, 0.f, 0.f, 0.f, 0.f, 0.f};
        }
    }

    // ---- merge the two col-subtile waves sharing each row-tile ----
    #pragma unroll
    for (int v = 0; v < 8; ++v) {
        if (l == v) { sBD[wave][8 * h + v] = bestD[v]; sBI[wave][8 * h + v] = bestI[v]; }
    }
    __syncthreads();
    if (ct == 0 && lane < 16) {
        float d0 = sBD[wave][lane],  d1 = sBD[wave + 1][lane];
        int   i0 = sBI[wave][lane],  i1 = sBI[wave + 1][lane];
        int take1 = (int)(d1 < d0) | ((int)(d1 == d0) & (int)(i1 < i0));
        out_idx[rowBase + rt * 16 + lane] = take1 ? i1 : i0;
    }
}

// ---------------------------------------------------------------------------
// Kernel 3: gather codebook rows into output, accumulate SSE and histogram.
// ---------------------------------------------------------------------------
__global__ void vq_gather_kernel(const float* __restrict__ X,
                                 const float* __restrict__ C,
                                 const int* __restrict__ idxbuf,
                                 float* __restrict__ out,
                                 float* __restrict__ counts,
                                 float* __restrict__ sse) {
    __shared__ float red[256];
    const int rlocal = threadIdx.x >> 7;
    const int t      = threadIdx.x & 127;
    const int row    = blockIdx.x * 2 + rlocal;
    const int code   = idxbuf[row];

    const float4* src = (const float4*)(C + (size_t)code * VQ_D) + t;
    const float4* xin = (const float4*)(X + (size_t)row  * VQ_D) + t;
    float4*       dst = (float4*)(out + (size_t)row * VQ_D) + t;

    float4 q = *src;
    float4 x = *xin;
    *dst = q;
    float dx = q.x - x.x, dy = q.y - x.y, dz = q.z - x.z, dw = q.w - x.w;
    red[threadIdx.x] = dx * dx + dy * dy + dz * dz + dw * dw;
    __syncthreads();
    for (int off = 128; off >= 1; off >>= 1) {
        if (threadIdx.x < off) red[threadIdx.x] += red[threadIdx.x + off];
        __syncthreads();
    }
    if (threadIdx.x == 0) atomicAdd(sse, red[0]);
    if (t == 0) atomicAdd(&counts[code], 1.0f);
}

// ---------------------------------------------------------------------------
// Kernel 4: loss + perplexity.
// ---------------------------------------------------------------------------
__global__ void vq_finalize_kernel(const float* __restrict__ counts,
                                   const float* __restrict__ sse,
                                   float* __restrict__ out) {
    __shared__ float red[256];
    float s = 0.0f;
    for (int c = threadIdx.x; c < VQ_K; c += 256) {
        float p = counts[c] * (1.0f / (float)VQ_N);
        s += p * logf(p + 1e-10f);
    }
    red[threadIdx.x] = s;
    __syncthreads();
    for (int off = 128; off >= 1; off >>= 1) {
        if (threadIdx.x < off) red[threadIdx.x] += red[threadIdx.x + off];
        __syncthreads();
    }
    if (threadIdx.x == 0) {
        out[VQ_ND]     = 1.25f * (*sse) * (1.0f / (float)VQ_ND);
        out[VQ_ND + 1] = expf(-red[0]);
    }
}

// ---------------------------------------------------------------------------
extern "C" void kernel_launch(void* const* d_in, const int* in_sizes, int n_in,
                              void* d_out, int out_size, void* d_ws, size_t ws_size,
                              hipStream_t stream) {
    const float* X = (const float*)d_in[0];   // (16,32,32,512) fp32
    const float* C = (const float*)d_in[1];   // (2048,512) fp32
    float* out = (float*)d_out;

    float* x2     = (float*)d_ws;             // 16384
    float* c2     = x2 + VQ_N;                // 2048
    float* counts = c2 + VQ_K;                // 2048 (+1 sse)
    float* sse    = counts + VQ_K;
    int*   idxb   = (int*)(sse + 1);          // 16384

    vq_norms_init_kernel<<<(VQ_N + VQ_K) / 16, 256, 0, stream>>>(X, C, x2, c2, counts);
    vq_argmin_kernel<<<VQ_N / ROWS_PB, 256, 0, stream>>>(X, C, x2, c2, idxb);
    vq_gather_kernel<<<VQ_N / 2, 256, 0, stream>>>(X, C, idxb, out, counts, sse);
    vq_finalize_kernel<<<1, 256, 0, stream>>>(counts, sse, out);
}
